// _TransformerAttention_26714696581664
// MI455X (gfx1250) — matrix operand order repaired and checked by reading, not compiled, not run on hardware
//
#include <hip/hip_runtime.h>
#include <hip/hip_bf16.h>

#define DEVINL __device__ __forceinline__

typedef __attribute__((ext_vector_type(16))) __bf16 v16bf;
typedef __attribute__((ext_vector_type(8)))  float  v8f;

namespace {

constexpr int Bc = 16;      // batch
constexpr int Tc = 1024;    // tokens
constexpr int Dc = 1024;    // model dim

constexpr int BM = 128;     // block tile M
constexpr int BN = 128;     // block tile N
constexpr int BK = 32;      // K chunk (one WMMA K)
constexpr int LDT = BK + 8; // padded LDS row (ushorts) to spread banks

DEVINL unsigned short f2bf(float x) {           // round-to-nearest-even fp32 -> bf16
  unsigned u = __float_as_uint(x);
  u += 0x7FFFu + ((u >> 16) & 1u);
  return (unsigned short)(u >> 16);
}

union Frag { v16bf v; uint4 q[2]; };

DEVINL v8f wmma_bf16(const Frag& a, const Frag& b, v8f c) {
  // D = A(16x32 bf16) * B(32x16 bf16) + C(f32)
  return __builtin_amdgcn_wmma_f32_16x16x32_bf16(false, a.v, false, b.v,
                                                 (short)0, c, false, false);
}

// CDNA5 async DMA: global -> LDS, 16B per lane, tracked by ASYNCcnt.
// VDST VGPR carries the per-lane LDS byte offset (low 32 bits of the generic
// pointer to a __shared__ object are the LDS offset on AMDGPU).
DEVINL void async_ld128(unsigned lds_off, const void* gaddr) {
  asm volatile("global_load_async_to_lds_b128 %0, %1, off"
               :: "v"(lds_off), "v"(gaddr) : "memory");
}
DEVINL void wait_async0() {
  asm volatile("s_wait_asynccnt 0" ::: "memory");
}

// MODE 0: f32 out, scaled by alpha.   MODE 1: bf16 out + bias.
// MODE 2: bf16 out + bias, written batch-transposed [b][n][t] (for V).
// A: [*, K] row-major (lda). Bt: [N, K] contraction-contiguous (ldb).
template<int MODE>
DEVINL void gemm_block(const unsigned short* __restrict__ A, int lda,
                       const unsigned short* __restrict__ Bt, int ldb,
                       int K, float alpha, const float* __restrict__ bias,
                       float* __restrict__ Cf, unsigned short* __restrict__ Cbf,
                       int ldc, int m0, int n0)
{
  __shared__ alignas(16) unsigned short As[2][BM * LDT];
  __shared__ alignas(16) unsigned short Bs[2][BN * LDT];

  const int tid   = threadIdx.x;       // 256 threads = 8 wave32
  const int lane  = tid & 31;
  const int wid   = tid >> 5;
  const int laneM = lane & 15;
  const int hi    = lane >> 4;         // half-wave select
  const int m_off = (wid & 3) * 32;    // 4 M-waves
  const int n_off = (wid >> 2) * 64;   // 2 N-waves

  const int lc = (tid & 3) * 8;        // k-column (8 bf16 = 16B) within tile
  const int lr = tid >> 2;             // tile row 0..63 (plus +64)

  // per-thread LDS destinations (byte offsets) for the 4 async transfers
  const unsigned dA0[2] = { (unsigned)(size_t)&As[0][(lr     ) * LDT + lc],
                            (unsigned)(size_t)&As[1][(lr     ) * LDT + lc] };
  const unsigned dA1[2] = { (unsigned)(size_t)&As[0][(lr + 64) * LDT + lc],
                            (unsigned)(size_t)&As[1][(lr + 64) * LDT + lc] };
  const unsigned dB0[2] = { (unsigned)(size_t)&Bs[0][(lr     ) * LDT + lc],
                            (unsigned)(size_t)&Bs[1][(lr     ) * LDT + lc] };
  const unsigned dB1[2] = { (unsigned)(size_t)&Bs[0][(lr + 64) * LDT + lc],
                            (unsigned)(size_t)&Bs[1][(lr + 64) * LDT + lc] };
  const unsigned short* gA0 = A  + (size_t)(m0 + lr     ) * lda + lc;
  const unsigned short* gA1 = A  + (size_t)(m0 + lr + 64) * lda + lc;
  const unsigned short* gB0 = Bt + (size_t)(n0 + lr     ) * ldb + lc;
  const unsigned short* gB1 = Bt + (size_t)(n0 + lr + 64) * ldb + lc;

  const v8f vzero = {0.f, 0.f, 0.f, 0.f, 0.f, 0.f, 0.f, 0.f};
  v8f acc[2][4];
#pragma unroll
  for (int s = 0; s < 2; ++s)
#pragma unroll
    for (int t = 0; t < 4; ++t) acc[s][t] = vzero;

  // prologue: DMA first K-chunk into buffer 0
  async_ld128(dA0[0], gA0);
  async_ld128(dA1[0], gA1);
  async_ld128(dB0[0], gB0);
  async_ld128(dB1[0], gB1);
  wait_async0();
  __syncthreads();

  const int nk = K / BK;
  for (int kc = 0; kc < nk; ++kc) {
    const int cur = kc & 1, nxt = cur ^ 1;
    const bool have_next = (kc + 1 < nk);
    if (have_next) {                       // kick off next-chunk DMA, overlap w/ WMMA
      const int k0 = (kc + 1) * BK;
      async_ld128(dA0[nxt], gA0 + k0);
      async_ld128(dA1[nxt], gA1 + k0);
      async_ld128(dB0[nxt], gB0 + k0);
      async_ld128(dB1[nxt], gB1 + k0);
    }
    if (kc + 2 < nk) {                     // L2 prefetch two chunks ahead
      const int kp = (kc + 2) * BK;
      __builtin_prefetch(gA0 + kp, 0, 3);
      __builtin_prefetch(gB0 + kp, 0, 3);
    }

    const unsigned short* Ab = As[cur];
    const unsigned short* Bb = Bs[cur];
    Frag af[2], bfr[4];
#pragma unroll
    for (int s = 0; s < 2; ++s) {          // A frag: two 16B runs per lane
      const unsigned short* p = Ab + (m_off + s * 16 + laneM) * LDT + hi * 8;
      af[s].q[0] = *(const uint4*)(p);
      af[s].q[1] = *(const uint4*)(p + 16);
    }
#pragma unroll
    for (int t = 0; t < 4; ++t) {          // B frag: two 16B runs per lane
      const unsigned short* p = Bb + (n_off + t * 16 + laneM) * LDT + hi * 8;
      bfr[t].q[0] = *(const uint4*)(p);
      bfr[t].q[1] = *(const uint4*)(p + 16);
    }
#pragma unroll
    for (int s = 0; s < 2; ++s)
#pragma unroll
      for (int t = 0; t < 4; ++t)
        acc[s][t] = wmma_bf16(af[s], bfr[t], acc[s][t]);

    if (have_next) wait_async0();          // our DMA into nxt has landed
    __syncthreads();                       // everyone's DMA visible
  }

  // epilogue: C/D layout -> row = e + hi*8 (+16*s), col = laneM (+16*t)
#pragma unroll
  for (int s = 0; s < 2; ++s) {
#pragma unroll
    for (int t = 0; t < 4; ++t) {
      const int cg    = n0 + n_off + t * 16 + laneM;
      const int rbase = m0 + m_off + s * 16 + hi * 8;
      float bb = 0.f;
      if (MODE != 0) bb = bias[cg];
      v8f c = acc[s][t];
#pragma unroll
      for (int e = 0; e < 8; ++e) {
        const int rg = rbase + e;
        if (MODE == 0) {
          Cf[(size_t)rg * ldc + cg] = c[e] * alpha;
        } else if (MODE == 1) {
          Cbf[(size_t)rg * ldc + cg] = f2bf(c[e] + bb);
        } else {                            // [b][n][t], T = 1024
          const int bidx = rg >> 10;
          const int tok  = rg & 1023;
          Cbf[((size_t)bidx * Dc + cg) * Tc + tok] = f2bf(c[e] + bb);
        }
      }
    }
  }
}

// ---------------- kernels ----------------

__global__ void __launch_bounds__(256)
k_cvt_x(const float* __restrict__ x, unsigned short* __restrict__ o) {
  const size_t i = (size_t)blockIdx.x * blockDim.x + threadIdx.x; // over n/4
  float4 v = ((const float4*)x)[i];
  ushort4 r;
  r.x = f2bf(v.x); r.y = f2bf(v.y); r.z = f2bf(v.z); r.w = f2bf(v.w);
  ((ushort4*)o)[i] = r;
}

__global__ void __launch_bounds__(256)
k_cvt_wt(const float* __restrict__ Wq, const float* __restrict__ Wk,
         const float* __restrict__ Wv, unsigned short* __restrict__ Wt) {
  __shared__ float tile[32][33];
  const int z = blockIdx.z;
  const float* W = (z == 0) ? Wq : (z == 1) ? Wk : Wv;
  const int k0 = blockIdx.x * 32, n0 = blockIdx.y * 32;
  const int tx = threadIdx.x & 31, ty = threadIdx.x >> 5;  // ty: 0..7
#pragma unroll
  for (int i = 0; i < 32; i += 8)
    tile[ty + i][tx] = W[(size_t)(k0 + ty + i) * Dc + n0 + tx];
  __syncthreads();
  unsigned short* out = Wt + (size_t)z * Dc * Dc;          // Wt[n][k]
#pragma unroll
  for (int i = 0; i < 32; i += 8)
    out[(size_t)(n0 + ty + i) * Dc + k0 + tx] = f2bf(tile[tx][ty + i]);
}

__global__ void __launch_bounds__(256)
k_proj_qk(const unsigned short* __restrict__ xbf, const unsigned short* __restrict__ Wt,
          const float* __restrict__ bq, const float* __restrict__ bk,
          unsigned short* __restrict__ Qbf, unsigned short* __restrict__ Kbf) {
  const int z = blockIdx.z;
  gemm_block<1>(xbf, Dc, Wt + (size_t)z * Dc * Dc, Dc, Dc, 1.0f,
                z == 0 ? bq : bk, nullptr, z == 0 ? Qbf : Kbf, Dc,
                blockIdx.x * BM, blockIdx.y * BN);
}

__global__ void __launch_bounds__(256)
k_proj_v(const unsigned short* __restrict__ xbf, const unsigned short* __restrict__ WtV,
         const float* __restrict__ bv, unsigned short* __restrict__ Vt) {
  gemm_block<2>(xbf, Dc, WtV, Dc, Dc, 1.0f, bv, nullptr, Vt, 0,
                blockIdx.x * BM, blockIdx.y * BN);
}

__global__ void __launch_bounds__(256)
k_scores(const unsigned short* __restrict__ Qbf, const unsigned short* __restrict__ Kbf,
         float* __restrict__ S) {
  const int b = blockIdx.z;
  gemm_block<0>(Qbf + (size_t)b * Tc * Dc, Dc, Kbf + (size_t)b * Tc * Dc, Dc,
                Dc, 0.03125f /* 1/sqrt(1024) */, nullptr,
                S + (size_t)b * Tc * Tc, nullptr, Tc,
                blockIdx.x * BM, blockIdx.y * BN);
}

// column-wise (axis=q) online max/sum: one thread per (b,k)
__global__ void __launch_bounds__(256)
k_softmax_stats(const float* __restrict__ S, float* __restrict__ mA,
                float* __restrict__ rZ) {
  const int idx = blockIdx.x * blockDim.x + threadIdx.x;   // 0..B*T-1
  const int b = idx >> 10, k = idx & 1023;
  const float* Sc = S + (size_t)b * Tc * Tc + k;
  float m = -3.402823466e38f, Z = 0.f;
  for (int q = 0; q < Tc; ++q) {
    const float v = Sc[(size_t)q * Tc];
    const float nm = fmaxf(m, v);
    Z = Z * __expf(m - nm) + __expf(v - nm);
    m = nm;
  }
  mA[idx] = m;
  rZ[idx] = 1.f / Z;
}

__global__ void __launch_bounds__(256)
k_softmax_apply(const float* __restrict__ S, const float* __restrict__ mA,
                const float* __restrict__ rZ, unsigned short* __restrict__ P) {
  const size_t i = (size_t)blockIdx.x * blockDim.x + threadIdx.x; // float4 index
  const size_t base = i * 4;
  const int b = (int)(base >> 20);          // T*T = 2^20
  const int k = (int)(base & 1023);
  const int mi = (b << 10) + k;
  float4 s4 = ((const float4*)S)[i];
  float4 m4 = *(const float4*)(mA + mi);
  float4 z4 = *(const float4*)(rZ + mi);
  ushort4 r;
  r.x = f2bf(__expf(s4.x - m4.x) * z4.x);
  r.y = f2bf(__expf(s4.y - m4.y) * z4.y);
  r.z = f2bf(__expf(s4.z - m4.z) * z4.z);
  r.w = f2bf(__expf(s4.w - m4.w) * z4.w);
  ((ushort4*)P)[i] = r;
}

__global__ void __launch_bounds__(256)
k_attn_out(const unsigned short* __restrict__ P, const unsigned short* __restrict__ Vt,
           float* __restrict__ O) {
  const int b = blockIdx.z;
  gemm_block<0>(P + (size_t)b * Tc * Tc, Tc, Vt + (size_t)b * Dc * Tc, Tc,
                Tc, 1.0f, nullptr, O + (size_t)b * Tc * Dc, nullptr, Dc,
                blockIdx.x * BM, blockIdx.y * BN);
}

__global__ void __launch_bounds__(256)
k_fc_partial(const float* __restrict__ O, const float* __restrict__ Wfc,
             float* __restrict__ part) {
  const int b = blockIdx.x, seg = blockIdx.y;              // 16 x 8
  const size_t segLen4 = (size_t)Tc * Dc / 8 / 4;          // 32768 float4
  const float4* Ob = (const float4*)(O + (size_t)b * Tc * Dc) + (size_t)seg * segLen4;
  const float4* Wp = (const float4*)Wfc + (size_t)seg * segLen4;
  float s = 0.f;
  for (size_t i = threadIdx.x; i < segLen4; i += 256) {
    float4 a = Ob[i], w = Wp[i];
    s += a.x * w.x + a.y * w.y + a.z * w.z + a.w * w.w;
  }
  __shared__ float red[256];
  red[threadIdx.x] = s;
  __syncthreads();
  for (int st = 128; st > 0; st >>= 1) {
    if ((int)threadIdx.x < st) red[threadIdx.x] += red[threadIdx.x + st];
    __syncthreads();
  }
  if (threadIdx.x == 0) part[b * 8 + seg] = red[0];
}

__global__ void k_fc_final(const float* __restrict__ part, const float* __restrict__ bfc,
                           float* __restrict__ y) {
  const int b = threadIdx.x;
  if (b < Bc) {
    float s = bfc[0];
    for (int j = 0; j < 8; ++j) s += part[b * 8 + j];
    y[b] = s;
  }
}

} // anonymous namespace

extern "C" void kernel_launch(void* const* d_in, const int* in_sizes, int n_in,
                              void* d_out, int out_size, void* d_ws, size_t ws_size,
                              hipStream_t stream) {
  (void)in_sizes; (void)n_in; (void)out_size; (void)ws_size;
  const float* x   = (const float*)d_in[0];
  const float* Wq  = (const float*)d_in[1];
  const float* bq  = (const float*)d_in[2];
  const float* Wk  = (const float*)d_in[3];
  const float* bk  = (const float*)d_in[4];
  const float* Wv  = (const float*)d_in[5];
  const float* bv  = (const float*)d_in[6];
  const float* Wfc = (const float*)d_in[7];
  const float* bfc = (const float*)d_in[8];
  float* y = (float*)d_out;

  // workspace layout (~198.2 MB); stream-ordered overlays: P reuses xbf, O reuses S
  char* ws = (char*)d_ws;
  const size_t MB = 1ull << 20;
  unsigned short* xbf = (unsigned short*)(ws);              // 32 MB (later: P)
  unsigned short* Pbf = xbf;
  unsigned short* Wt  = (unsigned short*)(ws + 32 * MB);    //  6 MB  [3][n][k]
  unsigned short* Qbf = (unsigned short*)(ws + 38 * MB);    // 32 MB
  unsigned short* Kbf = (unsigned short*)(ws + 70 * MB);    // 32 MB
  unsigned short* Vt  = (unsigned short*)(ws + 102 * MB);   // 32 MB  [b][d][t]
  float* S    = (float*)(ws + 134 * MB);                    // 64 MB (later: O)
  float* O    = S;
  float* mA   = (float*)(ws + 198 * MB);                    // 64 KB
  float* rZ   = (float*)(ws + 198 * MB + 64 * 1024);        // 64 KB
  float* part = (float*)(ws + 198 * MB + 128 * 1024);       // 512 B

  // 1) fp32 -> bf16 conversions (+ weight transpose for contraction-contiguous B)
  k_cvt_x<<<(Bc * Tc * Dc / 4) / 256, 256, 0, stream>>>(x, xbf);
  k_cvt_wt<<<dim3(Dc / 32, Dc / 32, 3), 256, 0, stream>>>(Wq, Wk, Wv, Wt);

  // 2) Q/K/V projections (V written pre-transposed per batch)
  k_proj_qk<<<dim3(Bc * Tc / BM, Dc / BN, 2), 256, 0, stream>>>(xbf, Wt, bq, bk, Qbf, Kbf);
  k_proj_v<<<dim3(Bc * Tc / BM, Dc / BN), 256, 0, stream>>>(xbf, Wt + (size_t)2 * Dc * Dc, bv, Vt);

  // 3) scores S = (Q K^T) / sqrt(D)  (fp32)
  k_scores<<<dim3(Tc / BM, Tc / BN, Bc), 256, 0, stream>>>(Qbf, Kbf, S);

  // 4) softmax over the QUERY axis (per-column stats, then normalize -> bf16 P)
  k_softmax_stats<<<(Bc * Tc) / 256, 256, 0, stream>>>(S, mA, rZ);
  k_softmax_apply<<<(size_t)Bc * Tc * Tc / 4 / 256, 256, 0, stream>>>(S, mA, rZ, Pbf);

  // 5) O = P V  (fp32 out, overlays S)
  k_attn_out<<<dim3(Tc / BM, Dc / BN, Bc), 256, 0, stream>>>(Pbf, Vt, O);

  // 6) final FC: y[b] = <O[b].flat, Wfc> + bfc
  k_fc_partial<<<dim3(Bc, 8), 256, 0, stream>>>(O, Wfc, part);
  k_fc_final<<<1, 32, 0, stream>>>(part, bfc, y);
}
